// HeteNet_72593537237024
// MI455X (gfx1250) — matrix operand order repaired and checked by reading, not compiled, not run on hardware
//
#include <hip/hip_runtime.h>

// ---------------------------------------------------------------------------
// Routed MoE MLP for MI455X (gfx1250, wave32, WMMA bf16).
//   out[t] = (relu(relu(obs[t] @ W1[e] + b1[e]) @ W2[e] + b2[e]) @ W3[e]) + b3[e]
// with e = agent_type[t]. Tokens are bucketed per expert (padded to 32 rows),
// activations/weights converted to bf16, three grouped GEMMs run with
// v_wmma_f32_16x16x32_bf16 (f32 accumulate), register-double-buffered.
// ---------------------------------------------------------------------------

typedef __attribute__((ext_vector_type(16))) __bf16 v16bf;
typedef __attribute__((ext_vector_type(8)))  __bf16 v8bf;
typedef __attribute__((ext_vector_type(8)))  float  v8f;

constexpr int TOK = 64 * 64;   // T*A = 4096 tokens
constexpr int F_  = 512;
constexpr int H_  = 1024;
constexpr int N_  = 512;
constexpr int E_  = 4;
constexpr int ROWCAP = TOK + E_ * 32;   // 4224 padded rows max (32-row pad)

// ---- workspace layout (bytes, all 64-aligned) ----------------------------
constexpr size_t OFF_CNT  = 0;
constexpr size_t OFF_FILL = 64;
constexpr size_t OFF_SEG  = 128;                       // seg[0..E_]
constexpr size_t OFF_IDX  = 256;                       // int idx[ROWCAP]
constexpr size_t OFF_XG   = 32768;                     // bf16 Xg [ROWCAP][F_]
constexpr size_t SZ_XG    = (size_t)ROWCAP * F_ * 2;
constexpr size_t OFF_H1   = OFF_XG + SZ_XG;            // bf16 H1 [ROWCAP][H_]
constexpr size_t SZ_H     = (size_t)ROWCAP * H_ * 2;
constexpr size_t OFF_H2   = OFF_H1 + SZ_H;             // bf16 H2 [ROWCAP][H_]
constexpr size_t OFF_W1T  = OFF_H2 + SZ_H;             // bf16 W1t [E][H_][F_]
constexpr size_t SZ_W1T   = (size_t)E_ * H_ * F_ * 2;
constexpr size_t OFF_W2T  = OFF_W1T + SZ_W1T;          // bf16 W2t [E][H_][H_]
constexpr size_t SZ_W2T   = (size_t)E_ * H_ * H_ * 2;
constexpr size_t OFF_W3T  = OFF_W2T + SZ_W2T;          // bf16 W3t [E][N_][H_]

__device__ __forceinline__ unsigned short f2bf(float f) {
    union { float f; unsigned u; } c; c.f = f;
    unsigned u = c.u;
    unsigned r = u + 0x7FFFu + ((u >> 16) & 1u);       // round-to-nearest-even
    return (unsigned short)(r >> 16);
}

// ---- routing --------------------------------------------------------------
__global__ void k_init(int* cnt, int* fill) {
    if (threadIdx.x < E_) { cnt[threadIdx.x] = 0; fill[threadIdx.x] = 0; }
}

__global__ void k_count(const int* __restrict__ at, int* __restrict__ cnt) {
    int t = blockIdx.x * blockDim.x + threadIdx.x;
    if (t < TOK) atomicAdd(&cnt[at[t]], 1);
}

__global__ void k_seg(const int* __restrict__ cnt, int* __restrict__ seg) {
    if (threadIdx.x == 0) {
        int o = 0;
        for (int e = 0; e < E_; ++e) {
            seg[e] = o;
            o += (cnt[e] + 31) & ~31;                  // pad each expert to 32 rows
        }
        seg[E_] = o;
    }
}

__global__ void k_fill(const int* __restrict__ at, const int* __restrict__ seg,
                       int* __restrict__ fill, int* __restrict__ idx) {
    int t = blockIdx.x * blockDim.x + threadIdx.x;
    if (t < TOK) {
        int e   = at[t];
        int pos = atomicAdd(&fill[e], 1);
        idx[seg[e] + pos] = t;
    }
}

// ---- gather obs rows -> bf16 Xg (zero-fill padded tail rows) --------------
__global__ void k_gather(const float* __restrict__ obs, const int* __restrict__ seg,
                         const int* __restrict__ cnt, const int* __restrict__ idx,
                         unsigned short* __restrict__ Xg) {
    int row = blockIdx.x;
    if (row >= seg[E_]) return;
    int e = 0;
    while (e < E_ - 1 && row >= seg[e + 1]) ++e;
    bool valid = (row - seg[e]) < cnt[e];
    int  tok   = valid ? idx[row] : 0;
    const float* src = obs + (size_t)tok * F_;
    unsigned short* dst = Xg + (size_t)row * F_;
    for (int f = threadIdx.x; f < F_; f += blockDim.x)
        dst[f] = valid ? f2bf(src[f]) : (unsigned short)0;
}

// ---- weight convert+transpose: src f32 (E,K,N) -> dst bf16 (E,N,K) --------
__global__ void k_cvt_t(const float* __restrict__ src, unsigned short* __restrict__ dst,
                        int K, int N) {
    size_t gid = (size_t)blockIdx.x * blockDim.x + threadIdx.x;
    size_t tot = (size_t)E_ * K * N;
    if (gid >= tot) return;
    int e   = (int)(gid / ((size_t)K * N));
    int rem = (int)(gid - (size_t)e * K * N);
    int k   = rem / N;
    int n   = rem - k * N;
    dst[((size_t)e * N + n) * K + k] = f2bf(src[gid]);
}

// ---- grouped GEMM: C[seg rows x NCOLS] = A @ B^T(e) + bias ----------------
// Block: 256 threads = 8 waves (4 M-waves x 2 N-waves). Block tile 128x128.
// Wave tile 32x64: 2 A-fragments, 4 shared B-fragments, 8 WMMA accumulators.
// Fragments are register-double-buffered across the K loop (step 32) so
// global_load_b128 for step k+32 overlaps the 8 WMMAs of step k.
struct Frags { v16bf a0, a1, b0, b1, b2, b3; };

template <int NCOLS, int KDIM, bool FINAL>
__global__ __launch_bounds__(256) void k_gemm(
        const unsigned short* __restrict__ A,      // bf16 [rows][KDIM], rows = global padded
        const unsigned short* __restrict__ BtAll,  // bf16 [E][NCOLS][KDIM] (transposed)
        const float* __restrict__ bias,            // f32 [E][NCOLS]
        unsigned short* __restrict__ Hout,         // bf16 [rows][NCOLS]   (!FINAL)
        float* __restrict__ out,                   // f32 [TOK][NCOLS]     (FINAL, scattered)
        const int* __restrict__ seg, const int* __restrict__ cnt,
        const int* __restrict__ idx) {
    int e    = blockIdx.z;
    int segS = seg[e];
    int len  = seg[e + 1] - segS;                  // multiple of 32
    int mB   = blockIdx.y * 128;
    if (mB >= len) return;

    int wave  = threadIdx.x >> 5;
    int lane  = threadIdx.x & 31;
    int waveM = wave & 3;
    int waveN = wave >> 2;
    int mTile = mB + waveM * 32;
    if (mTile >= len) return;                      // no LDS/barriers: safe early exit
    int nb = blockIdx.x * 128 + waveN * 64;

    const unsigned short* Bt = BtAll + (size_t)e * NCOLS * KDIM;

    // A fragment addressing (16-bit A 16x32 layout): lane&15 = row M,
    // lane>=16 holds the K+16..K+31 chunk (contiguous, matches B).
    int lo  = lane & 15;
    int hiA = (lane >> 4) << 3;                    // 0 or 8 halves
    int hiB = (lane >> 4) << 4;                    // 0 or 16 halves
    const unsigned short* Arow0 = A + (size_t)(segS + mTile + lo) * KDIM + hiB;
    const unsigned short* Arow1 = Arow0 + (size_t)16 * KDIM;
    const unsigned short* Bcol  = Bt + (size_t)(nb + lo) * KDIM + hiB;

    auto loadA16 = [](const unsigned short* p) -> v16bf {
        return *(const v16bf*)(p);                 // K..K+15 (or +16..+31)
    };
    auto loadFrags = [&](int k) -> Frags {
        Frags f;
        f.a0 = loadA16(Arow0 + k);
        f.a1 = loadA16(Arow1 + k);
        // B fragments: lane = column N, 16 contiguous K halves (32B) per lane.
        f.b0 = *(const v16bf*)(Bcol + k);
        f.b1 = *(const v16bf*)(Bcol + k + (size_t)16 * KDIM);
        f.b2 = *(const v16bf*)(Bcol + k + (size_t)32 * KDIM);
        f.b3 = *(const v16bf*)(Bcol + k + (size_t)48 * KDIM);
        return f;
    };

    v8f acc[2][4] = {};
    auto mma = [&](const Frags& f) {
#define WM(m, j, B)  acc[m][j] = __builtin_amdgcn_wmma_f32_16x16x32_bf16( \
        false, (m ? f.a1 : f.a0), false, f.B, (short)0, acc[m][j], false, false)
        WM(0, 0, b0); WM(1, 0, b0);
        WM(0, 1, b1); WM(1, 1, b1);
        WM(0, 2, b2); WM(1, 2, b2);
        WM(0, 3, b3); WM(1, 3, b3);
#undef WM
    };

    Frags cur = loadFrags(0);
    for (int k = 32; k < KDIM; k += 32) {
        Frags nxt = loadFrags(k);                  // overlaps with WMMAs below
        mma(cur);
        cur = nxt;
    }
    mma(cur);

    // C layout: VGPR r -> row M = r (+8 for lanes 16..31), lane&15 = col N.
#pragma unroll
    for (int m = 0; m < 2; ++m) {
        int rbase = segS + mTile + m * 16 + hiA;
#pragma unroll
        for (int j = 0; j < 4; ++j) {
            int   col = nb + j * 16 + lo;
            float bv  = bias[e * NCOLS + col];
#pragma unroll
            for (int r = 0; r < 8; ++r) {
                float v    = acc[m][j][r] + bv;
                int   grow = rbase + r;
                if (!FINAL) {
                    v = v > 0.f ? v : 0.f;                   // relu
                    Hout[(size_t)grow * NCOLS + col] = f2bf(v);
                } else {
                    if (grow - segS < cnt[e])                // skip pad rows
                        out[(size_t)idx[grow] * NCOLS + col] = v;
                }
            }
        }
    }
}

// ---------------------------------------------------------------------------
extern "C" void kernel_launch(void* const* d_in, const int* in_sizes, int n_in,
                              void* d_out, int out_size, void* d_ws, size_t ws_size,
                              hipStream_t stream) {
    const float* obs = (const float*)d_in[0];
    const float* W1  = (const float*)d_in[1];
    const float* b1  = (const float*)d_in[2];
    const float* W2  = (const float*)d_in[3];
    const float* b2  = (const float*)d_in[4];
    const float* W3  = (const float*)d_in[5];
    const float* b3  = (const float*)d_in[6];
    const int*   at  = (const int*)d_in[7];
    float*       out = (float*)d_out;

    char* ws = (char*)d_ws;
    int* cnt  = (int*)(ws + OFF_CNT);
    int* fill = (int*)(ws + OFF_FILL);
    int* seg  = (int*)(ws + OFF_SEG);
    int* idx  = (int*)(ws + OFF_IDX);
    unsigned short* Xg  = (unsigned short*)(ws + OFF_XG);
    unsigned short* H1  = (unsigned short*)(ws + OFF_H1);
    unsigned short* H2  = (unsigned short*)(ws + OFF_H2);
    unsigned short* W1t = (unsigned short*)(ws + OFF_W1T);
    unsigned short* W2t = (unsigned short*)(ws + OFF_W2T);
    unsigned short* W3t = (unsigned short*)(ws + OFF_W3T);

    // 1) routing
    k_init <<<1, 32, 0, stream>>>(cnt, fill);
    k_count<<<TOK / 256, 256, 0, stream>>>(at, cnt);
    k_seg  <<<1, 32, 0, stream>>>(cnt, seg);
    k_fill <<<TOK / 256, 256, 0, stream>>>(at, seg, fill, idx);

    // 2) bf16 conversion: gathered activations + transposed weights
    k_gather<<<ROWCAP, 128, 0, stream>>>(obs, seg, cnt, idx, Xg);
    {
        size_t n1 = (size_t)E_ * F_ * H_;   // W1: K=F_, N=H_
        size_t n2 = (size_t)E_ * H_ * H_;   // W2: K=H_, N=H_
        size_t n3 = (size_t)E_ * H_ * N_;   // W3: K=H_, N=N_
        k_cvt_t<<<(unsigned)((n1 + 255) / 256), 256, 0, stream>>>(W1, W1t, F_, H_);
        k_cvt_t<<<(unsigned)((n2 + 255) / 256), 256, 0, stream>>>(W2, W2t, H_, H_);
        k_cvt_t<<<(unsigned)((n3 + 255) / 256), 256, 0, stream>>>(W3, W3t, H_, N_);
    }

    // 3) grouped GEMMs (grid.y = worst-case M tiles, blocks early-exit on seg)
    dim3 blk(256);
    dim3 g1(H_ / 128, TOK / 128, E_);
    k_gemm<H_, F_, false><<<g1, blk, 0, stream>>>(Xg, W1t, b1, H1, nullptr, seg, cnt, idx);
    dim3 g2(H_ / 128, TOK / 128, E_);
    k_gemm<H_, H_, false><<<g2, blk, 0, stream>>>(H1, W2t, b2, H2, nullptr, seg, cnt, idx);
    dim3 g3(N_ / 128, TOK / 128, E_);
    k_gemm<N_, H_, true><<<g3, blk, 0, stream>>>(H2, W3t, b3, nullptr, out, seg, cnt, idx);
}
